// Attention_13709535609678
// MI455X (gfx1250) — matrix operand order repaired and checked by reading, not compiled, not run on hardware
//
#include <hip/hip_runtime.h>

// ---------------------------------------------------------------------------
// BEiT window attention for MI455X (gfx1250), bf16 WMMA everywhere.
// v3: software-pipelined GEMM k-loops (double-buffered fragments) so WMMAs
//     are not serialized behind s_wait_loadcnt 0; WGP-scope prefetch;
//     V-fragment loads hoisted under the softmax VALU shadow in attention.
// ---------------------------------------------------------------------------

#define DIMC   768
#define NHEAD  12
#define HD     64
#define NSEQ   1025
#define NPAD   1056          // 33 * 32 : flash loop steps evenly by 32 keys
#define BATCH  16
#define MTOT   (BATCH * NSEQ)   // 16400 = 1025 * 16 row-tiles exactly

typedef unsigned int  u32;
typedef unsigned short u16;
typedef __attribute__((ext_vector_type(4)))  u32     v4u;
typedef __attribute__((ext_vector_type(4)))  float   v4f;
typedef __attribute__((ext_vector_type(8)))  float   v8f;
typedef __attribute__((ext_vector_type(16))) __bf16  v16bf;

union Frag {
    v16bf v;
    v4u   q[2];
    u32   u[8];
};

__device__ inline u16 f2bf(float f) {
    u32 u = __float_as_uint(f);
    u32 r = u + 0x7FFFu + ((u >> 16) & 1u);   // round-to-nearest-even
    return (u16)(r >> 16);
}
__device__ inline u32 pk2(float a, float b) {
    return (u32)f2bf(a) | ((u32)f2bf(b) << 16);
}

// A-fragment (16x32 bf16): lane holds K {kb..kb+15} contiguous.
__device__ inline void load_a_bf16(Frag& f, const u16* p) {
    f.q[0] = *reinterpret_cast<const v4u*>(p);
    f.q[1] = *reinterpret_cast<const v4u*>(p + 8);
}
// B-fragment (32x16 bf16): lane holds K {kb..kb+15} contiguous.
__device__ inline void load_b_bf16(Frag& f, const u16* p) {
    f.q[0] = *reinterpret_cast<const v4u*>(p);
    f.q[1] = *reinterpret_cast<const v4u*>(p + 8);
}

__device__ inline v8f wmma_bf16(const Frag& a, const Frag& b, v8f c) {
    return __builtin_amdgcn_wmma_f32_16x16x32_bf16(
        false, a.v, false, b.v, (short)0, c, false, false);
}

// ---------------------------------------------------------------------------
// Kernel: bulk fp32 -> bf16 conversion (8 elems / thread, b128 in & out).
// ---------------------------------------------------------------------------
__global__ void cvt_bf16_kernel(const float* __restrict__ in,
                                u16* __restrict__ out, int n8) {
    int t = blockIdx.x * blockDim.x + threadIdx.x;
    if (t >= n8) return;
    const float* p = in + (size_t)t * 8;
    v4f f0 = *reinterpret_cast<const v4f*>(p);
    v4f f1 = *reinterpret_cast<const v4f*>(p + 4);
    v4u o;
    o.x = pk2(f0.x, f0.y); o.y = pk2(f0.z, f0.w);
    o.z = pk2(f1.x, f1.y); o.w = pk2(f1.z, f1.w);
    *reinterpret_cast<v4u*>(out + (size_t)t * 8) = o;
}

// ---------------------------------------------------------------------------
// Kernel: zero the pad rows of Q/K (row-major [bh][n][d]) and V^T ([bh][d][n])
// ---------------------------------------------------------------------------
__global__ void zero_pad_kernel(u16* __restrict__ qbuf, u16* __restrict__ kbuf,
                                u16* __restrict__ vbuf) {
    int tid = blockIdx.x * blockDim.x + threadIdx.x;
    const int PADR = NPAD - NSEQ;                 // 31
    const int total = BATCH * NHEAD * PADR * HD;
    if (tid >= total) return;
    int d  = tid & (HD - 1);
    int t  = tid >> 6;
    int row = NSEQ + (t % PADR);
    int bh  = t / PADR;
    qbuf[((size_t)bh * NPAD + row) * HD + d] = 0;
    kbuf[((size_t)bh * NPAD + row) * HD + d] = 0;
    vbuf[((size_t)bh * HD + d) * NPAD + row] = 0;
}

// ---------------------------------------------------------------------------
// Software-pipelined 16x64 GEMM core: two fragment buffers, loads for the
// next k-slice issued before the WMMAs of the current one.
// ---------------------------------------------------------------------------
__device__ inline void gemm_16x64(v8f acc[4], const u16* arow,
                                  const u16* const brow[4]) {
    Frag a0, a1, b0[4], b1[4];
    load_a_bf16(a0, arow);
    #pragma unroll
    for (int g = 0; g < 4; ++g) load_b_bf16(b0[g], brow[g]);

    for (int kk = 0; kk < DIMC; kk += 64) {
        __builtin_prefetch(arow + kk + 64, 0, 3);       // WGP-scope
        load_a_bf16(a1, arow + kk + 32);
        #pragma unroll
        for (int g = 0; g < 4; ++g) load_b_bf16(b1[g], brow[g] + kk + 32);
        #pragma unroll
        for (int g = 0; g < 4; ++g) acc[g] = wmma_bf16(a0, b0[g], acc[g]);

        if (kk + 64 < DIMC) {
            load_a_bf16(a0, arow + kk + 64);
            #pragma unroll
            for (int g = 0; g < 4; ++g) load_b_bf16(b0[g], brow[g] + kk + 64);
        }
        #pragma unroll
        for (int g = 0; g < 4; ++g) acc[g] = wmma_bf16(a1, b1[g], acc[g]);
    }
}

// ---------------------------------------------------------------------------
// Kernel: QKV projection.  out[m][n] = sum_k x[m][k] * qkv_w[n][k] (+bias)
// Each wave: 16 rows x 64 cols.  grid(1025, 9), block 128.
// ---------------------------------------------------------------------------
__global__ __launch_bounds__(128) void qkv_kernel(
    const u16* __restrict__ xbf, const u16* __restrict__ wbf,
    const float* __restrict__ q_bias, const float* __restrict__ v_bias,
    u16* __restrict__ qbuf, u16* __restrict__ kbuf, u16* __restrict__ vbuf) {

    const int lane = threadIdx.x & 31;
    const int wid  = threadIdx.x >> 5;
    const int la   = lane & 15;
    const int hi   = lane >> 4;
    const int mt   = blockIdx.x;
    const int ncol = (blockIdx.y * 4 + wid) * 64;

    v8f acc[4];
    #pragma unroll
    for (int g = 0; g < 4; ++g)
        #pragma unroll
        for (int r = 0; r < 8; ++r) acc[g][r] = 0.f;

    const int kbA = hi ? 16 : 0;
    const int kbB = hi ? 16 : 0;
    const u16* arow = xbf + (size_t)(mt * 16 + la) * DIMC + kbA;
    const u16* brow[4];
    #pragma unroll
    for (int g = 0; g < 4; ++g)
        brow[g] = wbf + (size_t)(ncol + g * 16 + la) * DIMC + kbB;

    gemm_16x64(acc, arow, brow);

    #pragma unroll
    for (int g = 0; g < 4; ++g) {
        const int col  = ncol + g * 16 + la;
        const int sect = col / DIMC;            // 0=q, 1=k, 2=v
        const int ch   = col - sect * DIMC;
        const int h    = ch >> 6;
        const int d    = ch & 63;
        #pragma unroll
        for (int r = 0; r < 8; ++r) {
            const int m   = mt * 16 + (hi ? r + 8 : r);
            const int b_  = m / NSEQ;
            const int row = m - b_ * NSEQ;
            const size_t bh = (size_t)(b_ * NHEAD + h);
            float v = acc[g][r];
            if (sect == 0) {
                v = (v + q_bias[ch]) * 0.125f;   // scale = 64^-0.5
                qbuf[(bh * NPAD + row) * HD + d] = f2bf(v);
            } else if (sect == 1) {
                kbuf[(bh * NPAD + row) * HD + d] = f2bf(v);
            } else {
                v += v_bias[col - 2 * DIMC];
                vbuf[(bh * HD + d) * NPAD + row] = f2bf(v);   // transposed V
            }
        }
    }
}

// ---------------------------------------------------------------------------
// Kernel: flash attention.  One wave per (b, h, 16-row q-tile).
// grid(65, 192), block 32.  Steps 32 keys at a time (2 x 16-col subtiles).
// ---------------------------------------------------------------------------
__global__ __launch_bounds__(32) void attn_kernel(
    const u16* __restrict__ qbuf, const u16* __restrict__ kbuf,
    const u16* __restrict__ vbuf, const float* __restrict__ rpb_table,
    const int* __restrict__ rel_index, u16* __restrict__ attn_out) {

    __shared__ u16 ptile[16 * 32];               // P transpose staging, 1 KB

    const int lane = threadIdx.x & 31;
    const int la   = lane & 15;
    const int hi   = lane >> 4;
    const int qt   = blockIdx.x;
    const int bh   = blockIdx.y;
    const int b_   = bh / NHEAD;
    const int h    = bh - b_ * NHEAD;

    const u16* qp = qbuf + ((size_t)bh * NPAD + qt * 16) * HD;
    const u16* kp = kbuf + (size_t)bh * NPAD * HD;
    const u16* vp = vbuf + (size_t)bh * HD * NPAD;

    // Q A-fragments for head-dim slices 0 and 1 (K=32 each)
    Frag qa[2];
    {
        const u16* p = qp + la * HD + (hi ? 16 : 0);
        load_a_bf16(qa[0], p);
        load_a_bf16(qa[1], p + 32);
    }

    v8f o[4];
    float mr[8], lr[8];
    #pragma unroll
    for (int r = 0; r < 8; ++r) {
        mr[r] = -__builtin_inff();
        lr[r] = 0.f;
        o[0][r] = 0.f; o[1][r] = 0.f; o[2][r] = 0.f; o[3][r] = 0.f;
    }

    const int kbB = hi ? 16 : 0;

    for (int jb = 0; jb < NPAD; jb += 32) {
        // ---- K and V fragment loads (V latency hides under softmax) ----
        const u16* k0 = kp + (size_t)(jb + la) * HD + kbB;
        const u16* k1 = kp + (size_t)(jb + 16 + la) * HD + kbB;
        Frag b00, b01, b10, b11, vb[4];
        load_b_bf16(b00, k0);        // subtile 0, head-dim slice 0
        load_b_bf16(b01, k0 + 32);   // subtile 0, slice 1
        load_b_bf16(b10, k1);
        load_b_bf16(b11, k1 + 32);
        #pragma unroll
        for (int g = 0; g < 4; ++g)
            load_b_bf16(vb[g], vp + (size_t)(g * 16 + la) * NPAD + jb + kbB);

        // ---- S = Q K^T for two 16-key subtiles -------------------------
        v8f s0 = {0.f,0.f,0.f,0.f,0.f,0.f,0.f,0.f};
        v8f s1 = {0.f,0.f,0.f,0.f,0.f,0.f,0.f,0.f};
        s0 = wmma_bf16(qa[0], b00, s0);
        s0 = wmma_bf16(qa[1], b01, s0);
        s1 = wmma_bf16(qa[0], b10, s1);
        s1 = wmma_bf16(qa[1], b11, s1);

        // ---- rel-pos bias, mask, online softmax ------------------------
        #pragma unroll
        for (int r = 0; r < 8; ++r) {
            const int row = qt * 16 + (hi ? r + 8 : r);
            const int c0  = jb + la;
            const int c1  = jb + 16 + la;
            float a0 = s0[r], a1 = s1[r];
            if (c0 < NSEQ) {
                if (row < NSEQ)
                    a0 += rpb_table[rel_index[row * NSEQ + c0] * NHEAD + h];
            } else a0 = -1e30f;
            if (c1 < NSEQ) {
                if (row < NSEQ)
                    a1 += rpb_table[rel_index[row * NSEQ + c1] * NHEAD + h];
            } else a1 = -1e30f;

            float tm = fmaxf(a0, a1);
            tm = fmaxf(tm, __shfl_xor(tm, 1, 32));
            tm = fmaxf(tm, __shfl_xor(tm, 2, 32));
            tm = fmaxf(tm, __shfl_xor(tm, 4, 32));
            tm = fmaxf(tm, __shfl_xor(tm, 8, 32));
            const float nm   = fmaxf(mr[r], tm);
            const float corr = __expf(mr[r] - nm);
            const float p0   = __expf(a0 - nm);
            const float p1   = __expf(a1 - nm);
            float ts = p0 + p1;
            ts += __shfl_xor(ts, 1, 32);
            ts += __shfl_xor(ts, 2, 32);
            ts += __shfl_xor(ts, 4, 32);
            ts += __shfl_xor(ts, 8, 32);
            lr[r] = lr[r] * corr + ts;
            mr[r] = nm;
            o[0][r] *= corr; o[1][r] *= corr; o[2][r] *= corr; o[3][r] *= corr;

            const int rloc = hi ? r + 8 : r;
            ptile[rloc * 32 + la]      = f2bf(p0);
            ptile[rloc * 32 + 16 + la] = f2bf(p1);
        }

        // ---- O += P V (C/D -> A layout via LDS) ------------------------
        Frag pa;
        load_a_bf16(pa, (const u16*)ptile + la * 32 + (hi ? 16 : 0));
        #pragma unroll
        for (int g = 0; g < 4; ++g)
            o[g] = wmma_bf16(pa, vb[g], o[g]);
    }

    // ---- normalize and write [B, N, H*64] bf16 -------------------------
    #pragma unroll
    for (int r = 0; r < 8; ++r) {
        const int row = qt * 16 + (hi ? r + 8 : r);
        if (row >= NSEQ) continue;
        const float inv = 1.0f / lr[r];
        const size_t base = ((size_t)b_ * NSEQ + row) * DIMC + h * HD;
        attn_out[base +      la] = f2bf(o[0][r] * inv);
        attn_out[base + 16 + la] = f2bf(o[1][r] * inv);
        attn_out[base + 32 + la] = f2bf(o[2][r] * inv);
        attn_out[base + 48 + la] = f2bf(o[3][r] * inv);
    }
}

// ---------------------------------------------------------------------------
// Kernel: output projection.  out = attn @ proj_w^T + proj_b  (fp32 out)
// grid(1025, 3), block 128.  Both streams bf16.
// ---------------------------------------------------------------------------
__global__ __launch_bounds__(128) void proj_kernel(
    const u16* __restrict__ attn, const u16* __restrict__ wbf,
    const float* __restrict__ proj_b, float* __restrict__ out) {

    const int lane = threadIdx.x & 31;
    const int wid  = threadIdx.x >> 5;
    const int la   = lane & 15;
    const int hi   = lane >> 4;
    const int mt   = blockIdx.x;
    const int ncol = (blockIdx.y * 4 + wid) * 64;

    v8f acc[4];
    #pragma unroll
    for (int g = 0; g < 4; ++g)
        #pragma unroll
        for (int r = 0; r < 8; ++r) acc[g][r] = 0.f;

    const int kbA = hi ? 16 : 0;
    const int kbB = hi ? 16 : 0;
    const u16* arow = attn + (size_t)(mt * 16 + la) * DIMC + kbA;
    const u16* brow[4];
    #pragma unroll
    for (int g = 0; g < 4; ++g)
        brow[g] = wbf + (size_t)(ncol + g * 16 + la) * DIMC + kbB;

    gemm_16x64(acc, arow, brow);

    #pragma unroll
    for (int g = 0; g < 4; ++g) {
        const int col = ncol + g * 16 + la;
        #pragma unroll
        for (int r = 0; r < 8; ++r) {
            const int m = mt * 16 + (hi ? r + 8 : r);
            out[(size_t)m * DIMC + col] = acc[g][r] + proj_b[col];
        }
    }
}

// ---------------------------------------------------------------------------
extern "C" void kernel_launch(void* const* d_in, const int* in_sizes, int n_in,
                              void* d_out, int out_size, void* d_ws, size_t ws_size,
                              hipStream_t stream) {
    (void)in_sizes; (void)n_in; (void)out_size; (void)ws_size;

    const float* x         = (const float*)d_in[0];
    const float* qkv_w     = (const float*)d_in[1];
    const float* q_bias    = (const float*)d_in[2];
    const float* v_bias    = (const float*)d_in[3];
    const float* rpb_table = (const float*)d_in[4];
    const float* proj_w    = (const float*)d_in[5];
    const float* proj_b    = (const float*)d_in[6];
    const int*   rel_index = (const int*)d_in[7];
    float*       out       = (float*)d_out;

    const size_t nqk    = (size_t)BATCH * NHEAD * NPAD * HD;  // q/k/v elems
    const size_t nx     = (size_t)MTOT * DIMC;                // x / attn elems
    const size_t nwqkv  = (size_t)3 * DIMC * DIMC;            // qkv_w elems
    const size_t nwproj = (size_t)DIMC * DIMC;                // proj_w elems

    u16* qbuf  = (u16*)d_ws;
    u16* kbuf  = qbuf  + nqk;
    u16* vbuf  = kbuf  + nqk;
    u16* attn  = vbuf  + nqk;      // [16400 x 768] bf16
    u16* xbf   = attn  + nx;       // [16400 x 768] bf16
    u16* wqkv  = xbf   + nx;       // [2304 x 768]  bf16
    u16* wproj = wqkv  + nwqkv;    // [768 x 768]   bf16

    // ---- one-shot fp32 -> bf16 conversions -----------------------------
    cvt_bf16_kernel<<<(int)((nx / 8 + 255) / 256), 256, 0, stream>>>(
        x, xbf, (int)(nx / 8));
    cvt_bf16_kernel<<<(int)((nwqkv / 8 + 255) / 256), 256, 0, stream>>>(
        qkv_w, wqkv, (int)(nwqkv / 8));
    cvt_bf16_kernel<<<(int)((nwproj / 8 + 255) / 256), 256, 0, stream>>>(
        proj_w, wproj, (int)(nwproj / 8));

    const int padElems = BATCH * NHEAD * (NPAD - NSEQ) * HD;
    zero_pad_kernel<<<(padElems + 255) / 256, 256, 0, stream>>>(qbuf, kbuf, vbuf);

    qkv_kernel<<<dim3(MTOT / 16, 9), 128, 0, stream>>>(
        xbf, wqkv, q_bias, v_bias, qbuf, kbuf, vbuf);

    attn_kernel<<<dim3((NSEQ + 15) / 16, BATCH * NHEAD), 32, 0, stream>>>(
        qbuf, kbuf, vbuf, rpb_table, rel_index, attn);

    proj_kernel<<<dim3(MTOT / 16, 3), 128, 0, stream>>>(
        attn, wproj, proj_b, out);
}
